// LSTM_adam_56315611186007
// MI455X (gfx1250) — matrix operand order repaired and checked by reading, not compiled, not run on hardware
//
#include <hip/hip_runtime.h>
#include <stdint.h>

// ---------------------------------------------------------------------------
// LSTM with Adam-like gate preconditioning, fused recurrent scan for gfx1250.
// B=32, T=1024, D=1024, H=1024, G=4H=4096.
//
// Per step t:  A = [x_t | h] (bf16), Wt = [W_ih^T ; W_hh^T] (4096x2048 bf16,
// L2-resident 16MB).  Block = 4 waves; wave w owns K quarter [512w, 512w+512)
// (waves 0-1 = x-part, waves 2-3 = h-part), 16-iteration WMMA chain with 4
// gate accumulators, LDS reduction, then lane-local Adam+LSTM pointwise.
// ---------------------------------------------------------------------------

typedef __bf16 bf16_t;
typedef __attribute__((ext_vector_type(16))) __bf16 v16bf;
typedef __attribute__((ext_vector_type(8)))  __bf16 v8bf;
typedef __attribute__((ext_vector_type(8)))  float  v8f;

#define B_  32
#define T_  1024
#define D_  1024
#define H_  1024
#define G_  4096
#define K2_ 2048   // D_ + H_

static constexpr float MU_   = 0.9f;
static constexpr float S_    = 0.01f;
static constexpr float BETA_ = 0.999f;
static constexpr float EPS_  = 1e-16f;

// d_out layout: y [B,T,H] | hT [B,H] | cT [B,H] | vT [B,G] | mT [B,G]
static constexpr size_t HOFF = (size_t)B_ * T_ * H_;
static constexpr size_t COFF = HOFF + (size_t)B_ * H_;
static constexpr size_t VOFF = COFF + (size_t)B_ * H_;
static constexpr size_t MOFF = VOFF + (size_t)B_ * G_;

__device__ __forceinline__ bf16_t f2bf(float f) {
  uint32_t u = __builtin_bit_cast(uint32_t, f);
  u += 0x7FFFu + ((u >> 16) & 1u);                 // round-to-nearest-even
  unsigned short s = (unsigned short)(u >> 16);
  return __builtin_bit_cast(bf16_t, s);
}

__device__ __forceinline__ float sigmoidf_(float x) {
  return 1.0f / (1.0f + __expf(-x));
}

// ---------------------------------------------------------------------------
// Prep: Wt[g][k] = bf16( k<1024 ? W_ih[k][g] : W_hh[k-1024][g] )  (transpose)
// ---------------------------------------------------------------------------
__global__ __launch_bounds__(256)
void prep_weights(const float* __restrict__ Wih,
                  const float* __restrict__ Whh,
                  bf16_t* __restrict__ Wt) {
  const int i = blockIdx.x * 256 + threadIdx.x;   // 0 .. 4096*2048-1
  const int g = i >> 11;
  const int k = i & (K2_ - 1);
  const float v = (k < D_) ? Wih[(size_t)k * G_ + g]
                           : Whh[(size_t)(k - D_) * G_ + g];
  Wt[i] = f2bf(v);
}

// ---------------------------------------------------------------------------
// Prep: xbf = bf16(x), 4 elements per thread (B*T*D / 4 threads total).
// ---------------------------------------------------------------------------
__global__ __launch_bounds__(256)
void prep_x(const float* __restrict__ x, bf16_t* __restrict__ xbf) {
  const size_t i = ((size_t)blockIdx.x * 256 + threadIdx.x) * 4;
  const float4 f = *(const float4*)(x + i);
  xbf[i + 0] = f2bf(f.x);
  xbf[i + 1] = f2bf(f.y);
  xbf[i + 2] = f2bf(f.z);
  xbf[i + 3] = f2bf(f.w);
}

// ---------------------------------------------------------------------------
// Init live state in d_out tail + bf16 shadow of h.
// ---------------------------------------------------------------------------
__global__ __launch_bounds__(256)
void init_state(const float* __restrict__ h0, const float* __restrict__ c0,
                const float* __restrict__ v0, const float* __restrict__ m0,
                float* __restrict__ out, bf16_t* __restrict__ hbf) {
  const int i = blockIdx.x * 256 + threadIdx.x;   // 0 .. B*G-1
  if (i < B_ * H_) {
    out[HOFF + i] = h0[i];
    out[COFF + i] = c0[i];
    hbf[i] = f2bf(h0[i]);
  }
  out[VOFF + i] = v0[i];
  out[MOFF + i] = m0[i];
}

// ---------------------------------------------------------------------------
// One recurrent step. 128 blocks x 4 waves. Block bid:
//   m0 = (bid&1)*16 (batch-row tile), n0 = (bid>>1)*16 (gate-column strip).
// Wave w reduces K in [512w, 512w+512); LDS combine; pointwise 2 rows/wave.
// ---------------------------------------------------------------------------
__global__ __launch_bounds__(128)
void lstm_step(const bf16_t* __restrict__ xbf,
               const float* __restrict__ bih,
               const float* __restrict__ bhh,
               const bf16_t* __restrict__ Wt,
               bf16_t* __restrict__ hbf,
               float* __restrict__ out,
               int t) {
  // red[(wave*4+gate)*384 + lane*12 + r] : stride 12 -> 16B-aligned b128
  // stores, <=2-way bank conflicts on the scalar reads.
  __shared__ float red[4 * 4 * 384];               // 24 KB

  const int tid  = threadIdx.x;
  const int wv   = tid >> 5;
  const int lane = tid & 31;
  const int half = lane >> 4;
  const int lq   = lane & 15;
  const int m0   = (blockIdx.x & 1) * 16;
  const int n0   = (blockIdx.x >> 1) * 16;

  const int rowA = m0 + lq;            // A-fragment row (batch index)
  const int ncol = n0 + lq;            // base gate column for this lane

  const int kbase = wv * 512;          // global K offset of this wave
  const bf16_t* arow =
      (wv < 2) ? (xbf + (size_t)rowA * (T_ * D_) + (size_t)t * D_ + kbase)
               : (hbf + (size_t)rowA * H_ + (kbase - D_));

  const bf16_t* w0 = Wt + (size_t)(ncol + 0 * H_) * K2_ + kbase + half * 16;
  const bf16_t* w1 = Wt + (size_t)(ncol + 1 * H_) * K2_ + kbase + half * 16;
  const bf16_t* w2 = Wt + (size_t)(ncol + 2 * H_) * K2_ + kbase + half * 16;
  const bf16_t* w3 = Wt + (size_t)(ncol + 3 * H_) * K2_ + kbase + half * 16;

  v8f acc[4] = {};

  for (int kk = 0; kk < 512; kk += 32) {
    const int d0 = kk + half * 16;
    const v8bf a0 = *(const v8bf*)(arow + d0);
    const v8bf a1 = *(const v8bf*)(arow + d0 + 8);
    const v16bf a = __builtin_shufflevector(a0, a1, 0, 1, 2, 3, 4, 5, 6, 7,
                                            8, 9, 10, 11, 12, 13, 14, 15);
    const v16bf b0 = *(const v16bf*)(w0 + kk);
    const v16bf b1 = *(const v16bf*)(w1 + kk);
    const v16bf b2 = *(const v16bf*)(w2 + kk);
    const v16bf b3 = *(const v16bf*)(w3 + kk);
    acc[0] = __builtin_amdgcn_wmma_f32_16x16x32_bf16(false, a, false, b0, (short)0, acc[0], false, false);
    acc[1] = __builtin_amdgcn_wmma_f32_16x16x32_bf16(false, a, false, b1, (short)0, acc[1], false, false);
    acc[2] = __builtin_amdgcn_wmma_f32_16x16x32_bf16(false, a, false, b2, (short)0, acc[2], false, false);
    acc[3] = __builtin_amdgcn_wmma_f32_16x16x32_bf16(false, a, false, b3, (short)0, acc[3], false, false);
  }

  // ---- stage partials to LDS ----------------------------------------------
#pragma unroll
  for (int g = 0; g < 4; ++g) {
    float* p = red + ((wv * 4 + g) * 384 + lane * 12);
    *(float4*)(p + 0) = make_float4(acc[g][0], acc[g][1], acc[g][2], acc[g][3]);
    *(float4*)(p + 4) = make_float4(acc[g][4], acc[g][5], acc[g][6], acc[g][7]);
  }
  __syncthreads();

  // ---- pointwise: wave w handles C-fragment rows r = 2w, 2w+1 -------------
  float* yv  = out;
  float* hst = out + HOFF;
  float* cst = out + COFF;
  float* vst = out + VOFF;
  float* mst = out + MOFF;

#pragma unroll
  for (int rr = 0; rr < 2; ++rr) {
    const int r    = 2 * wv + rr;
    const int brow = m0 + r + 8 * half;   // C/D layout: M = r + 8*(lane>>4)
    float gv[4];
#pragma unroll
    for (int g = 0; g < 4; ++g) {
      const int col = ncol + g * H_;
      const int e   = lane * 12 + r;
      const float sx = red[(0 * 4 + g) * 384 + e] + red[(1 * 4 + g) * 384 + e];
      const float sh = red[(2 * 4 + g) * 384 + e] + red[(3 * 4 + g) * 384 + e];
      const size_t sidx = (size_t)brow * G_ + col;
      const float gval = sx + bih[col];
      const float vN = MU_ * vst[sidx] + S_ * gval;
      const float mN = BETA_ * mst[sidx] + (1.0f - BETA_) * gval * gval;
      vst[sidx] = vN;
      mst[sidx] = mN;
      gv[g] = vN / (__builtin_sqrtf(mN) + EPS_) + sh + bhh[col];
    }
    const float iv = sigmoidf_(gv[0]);
    const float fv = sigmoidf_(gv[1]);
    const float cg = tanhf(gv[2]);
    const float ov = sigmoidf_(gv[3]);

    const size_t hidx = (size_t)brow * H_ + ncol;
    const float cN = cst[hidx] * fv + iv * cg;
    const float hN = ov * tanhf(cN);
    cst[hidx] = cN;
    hst[hidx] = hN;
    hbf[hidx] = f2bf(hN);
    yv[(size_t)brow * (T_ * H_) + (size_t)t * H_ + ncol] = hN;
  }
}

// ---------------------------------------------------------------------------
extern "C" void kernel_launch(void* const* d_in, const int* in_sizes, int n_in,
                              void* d_out, int out_size, void* d_ws, size_t ws_size,
                              hipStream_t stream) {
  const float* x   = (const float*)d_in[0];
  const float* h0  = (const float*)d_in[1];
  const float* c0  = (const float*)d_in[2];
  const float* v0  = (const float*)d_in[3];
  const float* m0  = (const float*)d_in[4];
  const float* Wih = (const float*)d_in[5];
  const float* Whh = (const float*)d_in[6];
  const float* bih = (const float*)d_in[7];
  const float* bhh = (const float*)d_in[8];

  float* out = (float*)d_out;
  bf16_t* Wt  = (bf16_t*)d_ws;                                      // 16 MB
  bf16_t* xbf = (bf16_t*)((char*)d_ws + (size_t)G_ * K2_ * 2);      // 64 MB
  bf16_t* hbf = xbf + (size_t)B_ * T_ * D_;                         // 64 KB

  prep_weights<<<(G_ * K2_) / 256, 256, 0, stream>>>(Wih, Whh, Wt);
  prep_x<<<((size_t)B_ * T_ * D_) / (256 * 4), 256, 0, stream>>>(x, xbf);
  init_state<<<(B_ * G_) / 256, 256, 0, stream>>>(h0, c0, v0, m0, out, hbf);

  for (int t = 0; t < T_; ++t) {
    lstm_step<<<128, 128, 0, stream>>>(xbf, bih, bhh, Wt, hbf, out, t);
  }
}
